// GNNEncoder_2362232013019
// MI455X (gfx1250) — matrix operand order repaired and checked by reading, not compiled, not run on hardware
//
#include <hip/hip_runtime.h>
#include <hip/hip_bf16.h>

typedef __attribute__((ext_vector_type(16))) __bf16 v16bf;
typedef __attribute__((ext_vector_type(8)))  float  v8f;

static __device__ __forceinline__ unsigned short f2bf(float f) {
    union { float f; unsigned u; } x; x.f = f;
    unsigned r = x.u + 0x7FFFu + ((x.u >> 16) & 1u);   // round-to-nearest-even
    return (unsigned short)(r >> 16);
}
static __device__ __forceinline__ unsigned pack2(float lo, float hi) {
    return (unsigned)f2bf(lo) | ((unsigned)f2bf(hi) << 16);
}

// ---------------------------------------------------------------------------
// hpre = (1 + eps) * x     (eps is a device scalar)
// ---------------------------------------------------------------------------
__global__ void init_hpre(const float* __restrict__ x, const float* __restrict__ eps,
                          float* __restrict__ hpre, long long n4) {
    long long i = (long long)blockIdx.x * blockDim.x + threadIdx.x;
    if (i >= n4) return;
    float s = 1.0f + *eps;
    float4 v = ((const float4*)x)[i];
    v.x *= s; v.y *= s; v.z *= s; v.w *= s;
    ((float4*)hpre)[i] = v;
}

// ---------------------------------------------------------------------------
// Edge messages: m = relu(x[src] + ea*We + be); atomic scatter-add into agg[dst]
// One thread = one edge x 4 features.  Gathers mostly hit the 192MB L2.
// ---------------------------------------------------------------------------
__global__ void edge_msg(const float* __restrict__ x, const int* __restrict__ src,
                         const int* __restrict__ dst, const float* __restrict__ ea,
                         const float* __restrict__ We, const float* __restrict__ be,
                         float* __restrict__ agg, int nedges, int c4n) {
    long long t = (long long)blockIdx.x * blockDim.x + threadIdx.x;
    long long total = (long long)nedges * c4n;
    if (t >= total) return;
    int e  = (int)(t / c4n);
    int c4 = (int)(t % c4n);
    int s = src[e], d = dst[e];
    float a = ea[e];
    float4 xv = ((const float4*)(x + (size_t)s * (c4n * 4)))[c4];
    float4 wv = ((const float4*)We)[c4];
    float4 bv = ((const float4*)be)[c4];
    float4 m;
    m.x = fmaxf(xv.x + a * wv.x + bv.x, 0.0f);
    m.y = fmaxf(xv.y + a * wv.y + bv.y, 0.0f);
    m.z = fmaxf(xv.z + a * wv.z + bv.z, 0.0f);
    m.w = fmaxf(xv.w + a * wv.w + bv.w, 0.0f);
    float* o = agg + (size_t)d * (c4n * 4) + c4 * 4;
    unsafeAtomicAdd(o + 0, m.x);
    unsafeAtomicAdd(o + 1, m.y);
    unsafeAtomicAdd(o + 2, m.z);
    unsafeAtomicAdd(o + 3, m.w);
}

// ---------------------------------------------------------------------------
// Pack f32 weights [K x Ncols] into bf16 WMMA B-operand tile layout.
// Tile (kt,nt) = 32x16; per lane L: n = nt*16 + (L&15), K-half base = (L&16)?8:0,
// VGPR v in 0..3: K = base+2v (lo), +1 (hi); v in 4..7: K = base+16+2(v-4).  Flat dword index:
//   ((nt*nkt + kt)*32 + lane)*8 + v
// ---------------------------------------------------------------------------
__global__ void pack_weights(const float* __restrict__ W, unsigned* __restrict__ P,
                             int K, int Ncols) {
    int tid = blockIdx.x * blockDim.x + threadIdx.x;
    int total = (K / 2) * Ncols;
    if (tid >= total) return;
    int t = tid >> 8;          // tile index = nt*nkt + kt
    int r = tid & 255;
    int lane = r >> 3, v = r & 7;
    int nkt = K >> 5;
    int kt = t % nkt, nt = t / nkt;
    int k0 = kt * 32 + ((lane & 16) ? 8 : 0) + ((v & 4) ? 16 : 0) + 2 * (v & 3);
    int n  = nt * 16 + (lane & 15);
    P[tid] = pack2(W[(size_t)k0 * Ncols + n], W[(size_t)(k0 + 1) * Ncols + n]);
}

// ---------------------------------------------------------------------------
// Pack f32 activations [M x K] into bf16 WMMA A-operand tile layout.
// Tile (mt,kt) = 16x32; per lane L: row = mt*16 + (L&15),
// K-half base = (L&16)?8:0; VGPR v in 0..3: K = base+2v; v in 4..7: K = base+16+2(v-4).
// Flat dword index: ((mt*nkt + kt)*32 + lane)*8 + v
// ---------------------------------------------------------------------------
__global__ void pack_act(const float* __restrict__ X, unsigned* __restrict__ P,
                         int K, long long total) {
    long long tid = (long long)blockIdx.x * blockDim.x + threadIdx.x;
    if (tid >= total) return;
    int t = (int)(tid >> 8);
    int r = (int)(tid & 255);
    int lane = r >> 3, v = r & 7;
    int nkt = K >> 5;
    int mt = t / nkt, kt = t % nkt;
    int row = mt * 16 + (lane & 15);
    int kb  = ((lane & 16) ? 8 : 0) + ((v & 4) ? 16 : 0);
    int k0  = kt * 32 + kb + 2 * (v & 3);
    const float* xr = X + (size_t)row * K + k0;
    P[tid] = pack2(xr[0], xr[1]);
}

// ---------------------------------------------------------------------------
// Fused GEMM (+bias) [+LayerNorm] +ReLU via v_wmma_f32_16x16x32_bf16.
// Both operands pre-packed in fragment layout: inner loop = 4x b128 load + wmma.
// Block = 512 threads = 16 waves; block computes 16 rows x 256 cols.
// ---------------------------------------------------------------------------
template <int K, int DO_LN>
__global__ __launch_bounds__(512) void gemm_fused(
    const unsigned* __restrict__ Apack, const unsigned* __restrict__ Wp,
    const float* __restrict__ bias, const float* __restrict__ gamma,
    const float* __restrict__ beta, float* __restrict__ Y) {
    __shared__ float sh[16 * 256];
    __shared__ float s_mu[16], s_rs[16];

    const int tid   = threadIdx.x;
    const int wave  = tid >> 5;
    const int lane  = tid & 31;
    const int mbase = blockIdx.x * 16;
    const int lo    = lane & 15;
    const int hiGrp = (lane >> 4) & 1;
    constexpr int NKT = K >> 5;

    v8f acc = {0.f, 0.f, 0.f, 0.f, 0.f, 0.f, 0.f, 0.f};
    const unsigned* abase = Apack + ((size_t)blockIdx.x * NKT * 32 + lane) * 8;
    const unsigned* bbase = Wp    + ((size_t)wave       * NKT * 32 + lane) * 8;

#pragma unroll
    for (int kt = 0; kt < NKT; ++kt) {
        union { unsigned u[8]; v16bf v; } A, B;
        const uint4* ap = (const uint4*)(abase + (size_t)kt * 256);
        const uint4* bp = (const uint4*)(bbase + (size_t)kt * 256);
        uint4 a0 = ap[0], a1 = ap[1];
        uint4 b0 = bp[0], b1 = bp[1];
        A.u[0] = a0.x; A.u[1] = a0.y; A.u[2] = a0.z; A.u[3] = a0.w;
        A.u[4] = a1.x; A.u[5] = a1.y; A.u[6] = a1.z; A.u[7] = a1.w;
        B.u[0] = b0.x; B.u[1] = b0.y; B.u[2] = b0.z; B.u[3] = b0.w;
        B.u[4] = b1.x; B.u[5] = b1.y; B.u[6] = b1.z; B.u[7] = b1.w;
        acc = __builtin_amdgcn_wmma_f32_16x16x32_bf16(
            false, A.v, false, B.v, (short)0, acc, false, false);
    }

    // C/D layout: VGPR r -> row r + 8*hiGrp, col = wave*16 + lo
    const int col = wave * 16 + lo;
    const float b = bias[col];
#pragma unroll
    for (int r = 0; r < 8; ++r)
        sh[(r + hiGrp * 8) * 256 + col] = acc[r] + b;
    __syncthreads();

    if (DO_LN) {
        // wave w reduces row w (256 values) for mean/var (wave32 shuffle tree)
        float s = 0.f, ss = 0.f;
#pragma unroll
        for (int j = 0; j < 8; ++j) {
            float v = sh[wave * 256 + lane + 32 * j];
            s += v; ss += v * v;
        }
#pragma unroll
        for (int off = 16; off > 0; off >>= 1) {
            s  += __shfl_xor(s, off, 32);
            ss += __shfl_xor(ss, off, 32);
        }
        if (lane == 0) {
            float mu  = s * (1.0f / 256.0f);
            float var = ss * (1.0f / 256.0f) - mu * mu;
            s_mu[wave] = mu;
            s_rs[wave] = rsqrtf(var + 1e-5f);
        }
        __syncthreads();
#pragma unroll
        for (int i = 0; i < 8; ++i) {
            int idx = tid + 512 * i;
            int row = idx >> 8, c = idx & 255;
            float v = (sh[idx] - s_mu[row]) * s_rs[row] * gamma[c] + beta[c];
            Y[(size_t)(mbase + row) * 256 + c] = fmaxf(v, 0.0f);
        }
    } else {
#pragma unroll
        for (int i = 0; i < 8; ++i) {
            int idx = tid + 512 * i;
            Y[(size_t)mbase * 256 + idx] = fmaxf(sh[idx], 0.0f);
        }
    }
}

// ---------------------------------------------------------------------------
extern "C" void kernel_launch(void* const* d_in, const int* in_sizes, int n_in,
                              void* d_out, int out_size, void* d_ws, size_t ws_size,
                              hipStream_t stream) {
    (void)in_sizes; (void)n_in; (void)out_size; (void)ws_size;
    const float* x    = (const float*)d_in[0];
    const int*   ei   = (const int*)  d_in[1];
    const float* ea   = (const float*)d_in[2];
    const float* eps1 = (const float*)d_in[3];
    const float* We1  = (const float*)d_in[4];
    const float* be1  = (const float*)d_in[5];
    const float* W11  = (const float*)d_in[6];
    const float* b11  = (const float*)d_in[7];
    const float* g1   = (const float*)d_in[8];
    const float* bb1  = (const float*)d_in[9];
    const float* W21  = (const float*)d_in[10];
    const float* b21  = (const float*)d_in[11];
    const float* eps2 = (const float*)d_in[12];
    const float* We2  = (const float*)d_in[13];
    const float* be2  = (const float*)d_in[14];
    const float* W12  = (const float*)d_in[15];
    const float* b12  = (const float*)d_in[16];
    const float* g2   = (const float*)d_in[17];
    const float* bb2  = (const float*)d_in[18];
    const float* W22  = (const float*)d_in[19];
    const float* b22  = (const float*)d_in[20];

    const int N = 50000, E = 800000, D = 128, H = 256;
    const int* src = ei;
    const int* dst = ei + E;

    char* ws = (char*)d_ws;
    size_t off = 0;
    auto alloc = [&](size_t bytes) -> char* {
        char* p = ws + off;
        off += (bytes + 255) & ~(size_t)255;
        return p;
    };
    float*    hpre1 = (float*)   alloc((size_t)N * D * 4);   // 25.6 MB
    float*    mid   = (float*)   alloc((size_t)N * H * 4);   // 51.2 MB
    float*    h1    = (float*)   alloc((size_t)N * H * 4);   // 51.2 MB
    float*    hpre2 = (float*)   alloc((size_t)N * H * 4);   // 51.2 MB
    unsigned* Apack = (unsigned*)alloc((size_t)N * H * 2);   // 25.6 MB (A staging)
    unsigned* W11p  = (unsigned*)alloc((size_t)D * H * 2);
    unsigned* W21p  = (unsigned*)alloc((size_t)H * H * 2);
    unsigned* W12p  = (unsigned*)alloc((size_t)H * H * 2);
    unsigned* W22p  = (unsigned*)alloc((size_t)H * H * 2);

    // --- pack weights into WMMA bf16 B-operand layout ---
    {
        int t1 = (D / 2) * H;
        int t2 = (H / 2) * H;
        pack_weights<<<(t1 + 255) / 256, 256, 0, stream>>>(W11, W11p, D, H);
        pack_weights<<<(t2 + 255) / 256, 256, 0, stream>>>(W21, W21p, H, H);
        pack_weights<<<(t2 + 255) / 256, 256, 0, stream>>>(W12, W12p, H, H);
        pack_weights<<<(t2 + 255) / 256, 256, 0, stream>>>(W22, W22p, H, H);
    }

    const long long packD = (long long)N * D / 2;   // dwords in A-pack, K=128
    const long long packH = (long long)N * H / 2;   // dwords in A-pack, K=256

    // --- layer 1 ---
    {
        long long n4 = (long long)N * D / 4;
        init_hpre<<<(unsigned)((n4 + 255) / 256), 256, 0, stream>>>(x, eps1, hpre1, n4);
        long long tot = (long long)E * (D / 4);
        edge_msg<<<(unsigned)((tot + 255) / 256), 256, 0, stream>>>(
            x, src, dst, ea, We1, be1, hpre1, E, D / 4);

        pack_act<<<(unsigned)((packD + 255) / 256), 256, 0, stream>>>(hpre1, Apack, D, packD);
        gemm_fused<128, 1><<<N / 16, 512, 0, stream>>>(Apack, W11p, b11, g1, bb1, mid);

        pack_act<<<(unsigned)((packH + 255) / 256), 256, 0, stream>>>(mid, Apack, H, packH);
        gemm_fused<256, 0><<<N / 16, 512, 0, stream>>>(Apack, W21p, b21, g1, bb1, h1);
    }
    // --- layer 2 ---
    {
        long long n4 = (long long)N * H / 4;
        init_hpre<<<(unsigned)((n4 + 255) / 256), 256, 0, stream>>>(h1, eps2, hpre2, n4);
        long long tot = (long long)E * (H / 4);
        edge_msg<<<(unsigned)((tot + 255) / 256), 256, 0, stream>>>(
            h1, src, dst, ea, We2, be2, hpre2, E, H / 4);

        pack_act<<<(unsigned)((packH + 255) / 256), 256, 0, stream>>>(hpre2, Apack, H, packH);
        gemm_fused<256, 1><<<N / 16, 512, 0, stream>>>(Apack, W12p, b12, g2, bb2, mid);

        pack_act<<<(unsigned)((packH + 255) / 256), 256, 0, stream>>>(mid, Apack, H, packH);
        gemm_fused<256, 0><<<N / 16, 512, 0, stream>>>(Apack, W22p, b22, g2, bb2,
                                                       (float*)d_out);
    }
}
